// MultiGATLayer_64106681860546
// MI455X (gfx1250) — matrix operand order repaired and checked by reading, not compiled, not run on hardware
//
#include <hip/hip_runtime.h>
#include <hip/hip_bf16.h>
#include <math.h>

#define N_NODES 4096
#define IN_DIM  512
#define OUT_DIM 64
#define NHEAD   8
#define EPSLN   1e-6f
#define LK      0.2f

typedef _Float16 v16h __attribute__((ext_vector_type(16)));
typedef _Float16 v8h  __attribute__((ext_vector_type(8)));
typedef float    v8f  __attribute__((ext_vector_type(8)));

union V16 { v16h v; v8h h[2]; };

// ---------------------------------------------------------------------------
// Kernel 0a: W (512x512 f32, row major) -> WhT (f16, transposed [col][row])
// ---------------------------------------------------------------------------
__global__ __launch_bounds__(256) void prep_wT(
    const float* __restrict__ W, _Float16* __restrict__ WhT)
{
    int idx = blockIdx.x * 256 + threadIdx.x;     // 262144 elements
    int r = idx >> 9, c = idx & 511;
    WhT[c * 512 + r] = (_Float16)W[idx];
}

// ---------------------------------------------------------------------------
// Kernel 0b: x (4096x512 f32) -> xh (f16, row major), coalesced convert
// ---------------------------------------------------------------------------
__global__ __launch_bounds__(256) void prep_xh(
    const float* __restrict__ x, _Float16* __restrict__ xh)
{
    int idx = blockIdx.x * 256 + threadIdx.x;     // 2M elements
    xh[idx] = (_Float16)x[idx];
}

// ---------------------------------------------------------------------------
// Kernel 1: nX = LayerNorm(x @ W)  (+ at1/at2 score vectors)
// 8 waves/block, wave w = head w, 32-node row block (2 A frags / B reuse x2).
// Staging: pure uint4 copies; wtT batched 8-deep for MLP before one wait.
// ---------------------------------------------------------------------------
__global__ __launch_bounds__(256) void gat_proj_ln(
    const _Float16* __restrict__ xh, const _Float16* __restrict__ WhT,
    const float* __restrict__ a1, const float* __restrict__ a2,
    const float* __restrict__ a_ln, const float* __restrict__ b_ln,
    float* __restrict__ nX, _Float16* __restrict__ nXhB,
    float* __restrict__ at1, float* __restrict__ at2)
{
    // main loop: xt 32x32 f16 (2KB) + wtT [512][32] f16 (32KB)
    // epilogue : stage 8x32x64 f32 (64KB)  -- unioned
    __shared__ __align__(16) unsigned char smem[8 * 32 * 64 * 4];
    _Float16* xt    = (_Float16*)smem;               // [32][32]
    uint4*    xt4   = (uint4*)smem;
    _Float16* wtT   = (_Float16*)(smem + 32*32*2);   // [512][32]
    uint4*    wtT4  = (uint4*)wtT;
    float*    stage = (float*)smem;                  // [8][32][64]

    const int tid   = threadIdx.x;
    const int wave  = tid >> 5;       // head index
    const int lane  = tid & 31;
    const int half_ = lane >> 4;
    const int l16   = lane & 15;
    const int rb    = blockIdx.x * 32;

    v8f acc[2][4] = {};
    const uint4* WhT4 = (const uint4*)WhT;           // col stride = 64 uint4
    const uint4* xh4  = (const uint4*)xh;            // row stride = 64 uint4

    for (int k0 = 0; k0 < IN_DIM; k0 += 32) {
        // x tile (32x32 f16): one b128 copy per thread (128 threads)
        if (tid < 128) {
            int row = tid >> 2, q = tid & 3;
            xt4[tid] = xh4[(size_t)(rb + row) * 64 + (k0 >> 3) + q];
        }
        // WhT tile (512 cols x 32 rows): 8 b128 per thread, batched
        {
            uint4 tmp[8];
            #pragma unroll
            for (int q = 0; q < 8; ++q) {
                int li = q * 256 + tid;              // 0..2047
                tmp[q] = WhT4[(size_t)(li >> 2) * 64 + (k0 >> 3) + (li & 3)];
            }
            #pragma unroll
            for (int q = 0; q < 8; ++q)
                wtT4[q * 256 + tid] = tmp[q];
        }
        __syncthreads();

        // two A fragments (row halves), each two contiguous b128 LDS loads
        V16 a0, a1f;
        a0.h[0]  = *(const v8h*)(xt + l16 * 32 + half_ * 8);
        a0.h[1]  = *(const v8h*)(xt + l16 * 32 + 16 + half_ * 8);
        a1f.h[0] = *(const v8h*)(xt + (16 + l16) * 32 + half_ * 8);
        a1f.h[1] = *(const v8h*)(xt + (16 + l16) * 32 + 16 + half_ * 8);

        #pragma unroll
        for (int t = 0; t < 4; ++t) {
            int cc = wave * 64 + t * 16 + l16;       // output column
            V16 b;
            b.h[0] = *(const v8h*)(wtT + cc * 32 + half_ * 8);
            b.h[1] = *(const v8h*)(wtT + cc * 32 + 16 + half_ * 8);
            acc[0][t] = __builtin_amdgcn_wmma_f32_16x16x32_f16(
                false, a0.v,  false, b.v, (short)0, acc[0][t], false, false);
            acc[1][t] = __builtin_amdgcn_wmma_f32_16x16x32_f16(
                false, a1f.v, false, b.v, (short)0, acc[1][t], false, false);
        }
        __syncthreads();
    }

    // dump accumulators to LDS stage (C layout: M = rh*16 + v + 8*half)
    float* st = stage + wave * 32 * 64;
    #pragma unroll
    for (int rh = 0; rh < 2; ++rh)
        #pragma unroll
        for (int t = 0; t < 4; ++t)
            #pragma unroll
            for (int v = 0; v < 8; ++v)
                st[(rh * 16 + v + half_ * 8) * 64 + t * 16 + l16] = acc[rh][t][v];
    __syncthreads();

    // LayerNorm epilogue: one node per lane (all 32 lanes busy)
    {
        int node = rb + lane;
        const float* row = st + lane * 64;
        float s = 0.f, sq = 0.f;
        #pragma unroll 8
        for (int o = 0; o < 64; ++o) { float v = row[o]; s += v; sq += v * v; }
        float mean = s * (1.f / 64.f);
        float var  = (sq - 64.f * mean * mean) * (1.f / 63.f);  // ddof=1
        var = var < 0.f ? 0.f : var;
        float inv = 1.f / (sqrtf(var) + EPSLN);
        float d1 = 0.f, d2 = 0.f;

        // nXhB fragment-order addressing for this (head, node)
        int jb = node >> 5, K = node & 31;
        int half_k = K >> 4, e = K & 15;
        size_t chunk = ((size_t)wave * 128 + jb) * 2048;
        size_t fbase = ((size_t)wave * N_NODES + node) * 64;
        #pragma unroll 8
        for (int o = 0; o < 64; ++o) {
            float v = a_ln[o] * (row[o] - mean) * inv + b_ln[o];
            nX[fbase + o] = v;
            int t = o >> 4, lane_b = half_k * 16 + (o & 15);
            nXhB[chunk + t * 512 + lane_b * 16 + e] = (_Float16)v;
            d1 += v * a1[o * NHEAD + wave];
            d2 += v * a2[o * NHEAD + wave];
        }
        at1[wave * N_NODES + node] = d1;
        at2[wave * N_NODES + node] = d2;
    }
}

// ---------------------------------------------------------------------------
// Kernel 2: per-(head,row) online softmax stats; one block per row,
// graph row streamed once for all 8 heads (memory-roofline pass).
// ---------------------------------------------------------------------------
__global__ __launch_bounds__(256) void gat_softmax_stats(
    const float* __restrict__ graph, const float* __restrict__ at1,
    const float* __restrict__ at2,
    float* __restrict__ rmax, float* __restrict__ rsum)
{
    __shared__ float smax[NHEAD][256];
    __shared__ float ssum[NHEAD][256];
    const int i   = blockIdx.x;
    const int tid = threadIdx.x;

    float a1v[NHEAD];
    #pragma unroll
    for (int k = 0; k < NHEAD; ++k) a1v[k] = at1[k * N_NODES + i];

    float m[NHEAD], sm[NHEAD];
    #pragma unroll
    for (int k = 0; k < NHEAD; ++k) { m[k] = -1e30f; sm[k] = 0.f; }

    const size_t rowbase = (size_t)i * N_NODES;
    for (int j = tid; j < N_NODES; j += 256) {
        __builtin_prefetch(&graph[rowbase + j + 2048], 0, 0); // global_prefetch_b8
        float g = graph[rowbase + j];
        if (g > 0.5f) {
            #pragma unroll
            for (int k = 0; k < NHEAD; ++k) {
                float s = a1v[k] + at2[k * N_NODES + j];
                s = s > 0.f ? s : LK * s;               // leaky_relu
                if (s > m[k]) { sm[k] = sm[k] * __expf(m[k] - s) + 1.f; m[k] = s; }
                else          { sm[k] += __expf(s - m[k]); }
            }
        }
    }
    #pragma unroll
    for (int k = 0; k < NHEAD; ++k) { smax[k][tid] = m[k]; ssum[k][tid] = sm[k]; }
    __syncthreads();

    for (int off = 128; off > 0; off >>= 1) {
        if (tid < off) {
            #pragma unroll
            for (int k = 0; k < NHEAD; ++k) {
                float ma = smax[k][tid], mb = smax[k][tid + off];
                float sa = ssum[k][tid], sb = ssum[k][tid + off];
                float mm = ma > mb ? ma : mb;
                smax[k][tid] = mm;
                ssum[k][tid] = sa * __expf(ma - mm) + sb * __expf(mb - mm);
            }
        }
        __syncthreads();
    }
    if (tid == 0) {
        #pragma unroll
        for (int k = 0; k < NHEAD; ++k) {
            rmax[k * N_NODES + i] = smax[k][0];
            rsum[k * N_NODES + i] = ssum[k][0];
        }
    }
}

// ---------------------------------------------------------------------------
// Kernel 3: y = softmax(P) @ nX + nX, elu.  32-row blocks: two P A-fragments
// per wave share each B fragment (global_load_b128 from fragment-ordered
// nXhB, L2-resident) -> 8 WMMA per staged tile.
// ---------------------------------------------------------------------------
__global__ __launch_bounds__(256) void gat_attn_mm(
    const float* __restrict__ graph, const float* __restrict__ at1,
    const float* __restrict__ at2,  const float* __restrict__ rmax,
    const float* __restrict__ rsum, const float* __restrict__ nX,
    const _Float16* __restrict__ nXhB, float* __restrict__ out)
{
    __shared__ __align__(16) float gt[32 * 32]; // graph tile, 4KB
    __shared__ float a2t[NHEAD * 32];           // at2 chunk, 1KB
    float4* gt4 = (float4*)gt;

    const int tid   = threadIdx.x;
    const int wave  = tid >> 5;    // head
    const int lane  = tid & 31;
    const int half_ = lane >> 4;
    const int l16   = lane & 15;
    const int rb    = blockIdx.x * 32;

    // per-lane row scalars for the two A-fragment row halves
    const float a1v0 = at1[wave * N_NODES + rb + l16];
    const float a1v1 = at1[wave * N_NODES + rb + 16 + l16];
    // fold 1/Z into the exponent: p = exp(s - (max + log Z))
    const float lse0 = rmax[wave * N_NODES + rb + l16]
                     + __logf(rsum[wave * N_NODES + rb + l16]);
    const float lse1 = rmax[wave * N_NODES + rb + 16 + l16]
                     + __logf(rsum[wave * N_NODES + rb + 16 + l16]);

    v8f acc[2][4] = {};
    const float4* graph4 = (const float4*)graph;     // row stride = 1024 float4

    for (int j0 = 0; j0 < N_NODES; j0 += 32) {
        // graph tile 32x32: one b128 copy per thread (+ prefetch next tile)
        {
            int row = tid >> 3, c4 = tid & 7;
            size_t idx4 = (size_t)(rb + row) * 1024 + (j0 >> 2) + c4;
            __builtin_prefetch(&graph4[idx4 + 8], 0, 0);   // next 32-col tile
            gt4[tid] = graph4[idx4];
        }
        // at2 chunk for all heads
        if (tid < NHEAD * 32) {
            int k = tid >> 5, jj = tid & 31;
            a2t[tid] = at2[k * N_NODES + j0 + jj];
        }
        __syncthreads();

        // two A fragments = probabilities P[M, j0+K] in f16 (VALU-built)
        V16 a0, a1f;
        #pragma unroll
        for (int e = 0; e < 16; ++e) {
            int K = e + half_ * 16;
            float s2 = a2t[wave * 32 + K];
            float g0 = gt[l16 * 32 + K];
            float g1 = gt[(16 + l16) * 32 + K];
            float s0 = a1v0 + s2;  s0 = s0 > 0.f ? s0 : LK * s0;
            float s1 = a1v1 + s2;  s1 = s1 > 0.f ? s1 : LK * s1;
            a0.v[e]  = (_Float16)((g0 > 0.5f) ? __expf(s0 - lse0) : 0.f);
            a1f.v[e] = (_Float16)((g1 > 0.5f) ? __expf(s1 - lse1) : 0.f);
        }

        // B fragments from global nXhB (fragment order, 2x b128), reused x2
        const _Float16* chunk =
            nXhB + ((size_t)wave * 128 + (j0 >> 5)) * 2048 + lane * 16;
        #pragma unroll
        for (int t = 0; t < 4; ++t) {
            V16 b;
            b.h[0] = *(const v8h*)(chunk + t * 512);
            b.h[1] = *(const v8h*)(chunk + t * 512 + 8);
            acc[0][t] = __builtin_amdgcn_wmma_f32_16x16x32_f16(
                false, a0.v,  false, b.v, (short)0, acc[0][t], false, false);
            acc[1][t] = __builtin_amdgcn_wmma_f32_16x16x32_f16(
                false, a1f.v, false, b.v, (short)0, acc[1][t], false, false);
        }
        __syncthreads();
    }

    // epilogue: residual + elu, scatter to (node, head*64+n)
    #pragma unroll
    for (int rh = 0; rh < 2; ++rh)
        #pragma unroll
        for (int t = 0; t < 4; ++t)
            #pragma unroll
            for (int v = 0; v < 8; ++v) {
                int node = rb + rh * 16 + v + half_ * 8;
                int n = t * 16 + l16;
                float val = acc[rh][t][v]
                          + nX[((size_t)wave * N_NODES + node) * 64 + n];
                val = val > 0.f ? val : (__expf(val) - 1.f);   // elu
                out[(size_t)node * (NHEAD * OUT_DIM) + wave * OUT_DIM + n] = val;
            }
}

// ---------------------------------------------------------------------------
extern "C" void kernel_launch(void* const* d_in, const int* in_sizes, int n_in,
                              void* d_out, int out_size, void* d_ws, size_t ws_size,
                              hipStream_t stream) {
    (void)in_sizes; (void)n_in; (void)out_size; (void)ws_size;
    const float* x     = (const float*)d_in[0];
    const float* graph = (const float*)d_in[1];
    // d_in[2] = mask (bool) — adjacency recovered from `graph` (0/1 floats)
    const float* W     = (const float*)d_in[3];
    const float* a1    = (const float*)d_in[4];
    const float* a2    = (const float*)d_in[5];
    const float* a_ln  = (const float*)d_in[6];
    const float* b_ln  = (const float*)d_in[7];
    float* out = (float*)d_out;

    char* ws = (char*)d_ws;
    float*    nX   = (float*)ws;                               // 8 MiB
    _Float16* nXhB = (_Float16*)(ws + 8u * 1024 * 1024);       // 4 MiB (frag order)
    float*    at1  = (float*)(ws + 12u * 1024 * 1024);         // 4 x 128 KiB
    float*    at2  = at1 + NHEAD * N_NODES;
    float*    rmx  = at2 + NHEAD * N_NODES;
    float*    rsm  = rmx + NHEAD * N_NODES;
    _Float16* WhT  = (_Float16*)(ws + 12u * 1024 * 1024 + 512u * 1024); // 512 KiB
    _Float16* xh   = (_Float16*)(ws + 13u * 1024 * 1024);      // 4 MiB

    prep_wT          <<<(IN_DIM * IN_DIM) / 256, 256, 0, stream>>>(W, WhT);
    prep_xh          <<<(N_NODES * IN_DIM) / 256, 256, 0, stream>>>(x, xh);
    gat_proj_ln      <<<N_NODES / 32, 256, 0, stream>>>(xh, WhT, a1, a2, a_ln, b_ln,
                                                        nX, nXhB, at1, at2);
    gat_softmax_stats<<<N_NODES,      256, 0, stream>>>(graph, at1, at2, rmx, rsm);
    gat_attn_mm      <<<N_NODES / 32, 256, 0, stream>>>(graph, at1, at2, rmx, rsm,
                                                        nX, nXhB, out);
}
